// PatchedMHA_68504728371382
// MI455X (gfx1250) — compile-verified, matrix operand order checked by reading, not run on hardware
//
#include <hip/hip_runtime.h>
#include <math.h>

typedef __attribute__((ext_vector_type(16))) _Float16 v16h;
typedef __attribute__((ext_vector_type(8)))  _Float16 v8h;
typedef __attribute__((ext_vector_type(4)))  _Float16 v4h;
typedef __attribute__((ext_vector_type(8)))  float    v8f;
typedef __attribute__((ext_vector_type(4)))  int      v4i;

#define B_  2
#define S_  2048
#define D_  1024
#define H_  16
#define DK_ 64
#define M_  (B_ * S_)   // 4096
#define K_  D_          // 1024
#define N_  D_          // 1024

#define BM 128
#define BN 128
#define BK 32
#define LDA 40          // padded LDS row stride in halves (80B, keeps 16B chunk alignment)

// ---- CDNA5 async global->LDS path (guarded: falls back to sync staging) ----
#if defined(__has_builtin)
#if __has_builtin(__builtin_amdgcn_global_load_async_to_lds_b128)
#define HAVE_ASYNC_LDS 1
#endif
#endif
#ifndef HAVE_ASYNC_LDS
#define HAVE_ASYNC_LDS 0
#endif

__device__ __forceinline__ void wait_async0() {
#if defined(__has_builtin) && __has_builtin(__builtin_amdgcn_s_wait_asynccnt)
  __builtin_amdgcn_s_wait_asynccnt(0);
#else
  asm volatile("s_wait_asynccnt 0x0" ::: "memory");
#endif
}

__device__ __forceinline__ void async_copy_b128(const _Float16* gsrc, _Float16* ldst) {
#if HAVE_ASYNC_LDS
  // Builtin takes generic v4i* pointers (probe-confirmed param type).
  __builtin_amdgcn_global_load_async_to_lds_b128((v4i*)gsrc, (v4i*)ldst, 0, 0);
#else
  *(v8h*)ldst = *(const v8h*)gsrc;
#endif
}

__device__ __forceinline__ v16h make_a16(v8h lo, v8h hi) {
  v16h a;
#pragma unroll
  for (int j = 0; j < 8; ++j) { a[j] = lo[j]; a[8 + j] = hi[j]; }
  return a;
}

// A/B fragment for v_wmma_f32_16x16x32_f16 from a row-major (stride-LDA) row.
// Lane l: K chunks [kb8, kb8+8) and [kb8+16, kb8+24), kb8 = (lane>>4)*8.
__device__ __forceinline__ v16h frag_from_row(const _Float16* rowbase, int kb8) {
  v8h lo = *(const v8h*)(rowbase + kb8);
  v8h hi = *(const v8h*)(rowbase + kb8 + 16);
  return make_a16(lo, hi);
}

// ---------------------------------------------------------------------------
// f32 -> f16 conversion pass (query/key/value); blockIdx.y selects tensor.
// ---------------------------------------------------------------------------
__global__ __launch_bounds__(256) void cvt_f32_to_f16(
    const float* __restrict__ x0, const float* __restrict__ x1,
    const float* __restrict__ x2, _Float16* __restrict__ y0,
    _Float16* __restrict__ y1, _Float16* __restrict__ y2) {
  const float* x = (blockIdx.y == 0) ? x0 : (blockIdx.y == 1) ? x1 : x2;
  _Float16*    y = (blockIdx.y == 0) ? y0 : (blockIdx.y == 1) ? y1 : y2;
  const int i = blockIdx.x * 256 + threadIdx.x;  // over float4 chunks
  const float4 v = ((const float4*)x)[i];
  v4h o;
  o[0] = (_Float16)v.x; o[1] = (_Float16)v.y;
  o[2] = (_Float16)v.z; o[3] = (_Float16)v.w;
  ((v4h*)y)[i] = o;
}

// ---------------------------------------------------------------------------
// GEMM: out = A[M,K](f16) * W[K,N](f32->f16) + bias, f32 accumulate.
// OUT_MODE: 0 = f32 [M,N] | 1 = f16 [B,H,S,DK] (Q,K) | 2 = f16 [B,H,DK,S] (V)
// Double-buffered LDS; A tile DMA'd via GLOBAL_LOAD_ASYNC_TO_LDS_B128.
// ---------------------------------------------------------------------------
template <int OUT_MODE>
__global__ __launch_bounds__(256) void mha_gemm_wmma(
    const _Float16* __restrict__ A, const float* __restrict__ W,
    const float* __restrict__ bias, void* __restrict__ out) {
  __shared__ __align__(16) _Float16 ldsA[2][BM][LDA];
  __shared__ __align__(16) _Float16 ldsW[2][BN][LDA];

  const int tid  = threadIdx.x;
  const int lane = tid & 31;
  const int wave = tid >> 5;
  const int wm   = wave >> 1;   // 0..3 -> 32-row strip
  const int wn   = wave & 1;    // 0..1 -> 64-col strip
  const int m0   = blockIdx.y * BM;
  const int n0   = blockIdx.x * BN;
  const int l15  = lane & 15;
  const int half = lane >> 4;
  const int kb8  = half * 8;

  auto stageA = [&](int buf, int k0) {
    // BM x BK halves = 512 16B chunks, 2 per thread
#pragma unroll
    for (int i = 0; i < 2; ++i) {
      const int idx = tid + i * 256;          // 0..511
      const int r = idx >> 2, c8 = (idx & 3) * 8;
      async_copy_b128(A + (size_t)(m0 + r) * K_ + k0 + c8, &ldsA[buf][r][c8]);
    }
  };
  auto stageW = [&](int buf, int k0) {  // f32->f16 + transpose, VALU path
    for (int i = tid; i < BK * BN; i += 256) {
      const int kk = i >> 7, nn = i & 127;
      ldsW[buf][nn][kk] = (_Float16)W[(size_t)(k0 + kk) * N_ + n0 + nn];
    }
    if (k0 + BK < K_)  // hint next W tile (global_prefetch_b8)
      __builtin_prefetch(&W[(size_t)(k0 + BK + (tid >> 3)) * N_ + n0 + (tid & 7) * 16], 0, 1);
  };

  v8f zero = {};
  v8f acc[2][4];
#pragma unroll
  for (int i = 0; i < 2; ++i)
#pragma unroll
    for (int j = 0; j < 4; ++j) acc[i][j] = zero;

  // Prologue: fill buffer 0
  stageA(0, 0);
  stageW(0, 0);
#if HAVE_ASYNC_LDS
  wait_async0();
#endif
  __syncthreads();

  int buf = 0;
  for (int k0 = 0; k0 < K_; k0 += BK, buf ^= 1) {
    const int nk = k0 + BK;
    if (nk < K_) stageA(buf ^ 1, nk);  // async DMA overlaps with compute below

    v16h afrag[2];
#pragma unroll
    for (int mi = 0; mi < 2; ++mi)
      afrag[mi] = frag_from_row(&ldsA[buf][wm * 32 + mi * 16 + l15][0], kb8);
#pragma unroll
    for (int ni = 0; ni < 4; ++ni) {
      v16h bfrag = frag_from_row(&ldsW[buf][wn * 64 + ni * 16 + l15][0], kb8);
#pragma unroll
      for (int mi = 0; mi < 2; ++mi)
        acc[mi][ni] = __builtin_amdgcn_wmma_f32_16x16x32_f16(
            false, afrag[mi], false, bfrag, (short)0, acc[mi][ni], false, false);
    }

    if (nk < K_) stageW(buf ^ 1, nk);
#if HAVE_ASYNC_LDS
    wait_async0();
#endif
    __syncthreads();
  }

  // Epilogue: bias add + layout-specific store
#pragma unroll
  for (int mi = 0; mi < 2; ++mi) {
#pragma unroll
    for (int ni = 0; ni < 4; ++ni) {
      const int n  = n0 + wn * 64 + ni * 16 + l15;
      const float bn = bias[n];
#pragma unroll
      for (int r = 0; r < 8; ++r) {
        const int m = m0 + wm * 32 + mi * 16 + r + 8 * half;
        float v = acc[mi][ni][r] + bn;
        if constexpr (OUT_MODE == 0) {
          ((float*)out)[(size_t)m * N_ + n] = v;
        } else {
          const int b = m >> 11, s = m & (S_ - 1);
          const int h = n >> 6, dk = n & (DK_ - 1);
          size_t dst;
          if constexpr (OUT_MODE == 1)
            dst = ((size_t)(b * H_ + h) * S_ + s) * DK_ + dk;
          else
            dst = ((size_t)(b * H_ + h) * DK_ + dk) * S_ + s;
          ((_Float16*)out)[dst] = (_Float16)v;
        }
      }
    }
  }
}

// ---------------------------------------------------------------------------
// Flash attention: one wave32 per 16-query tile; online softmax over 32-key
// blocks; scores never touch HBM.  Q,K: f16 [B,H,S,DK]; V: f16 [B,H,DK,S].
// ---------------------------------------------------------------------------
__global__ __launch_bounds__(256) void mha_flash_wmma(
    const _Float16* __restrict__ Q, const _Float16* __restrict__ Kf,
    const _Float16* __restrict__ Vt, const unsigned char* __restrict__ mask,
    _Float16* __restrict__ O) {
  __shared__ __align__(16) _Float16 ldsP[8][16][LDA];  // per-wave 16x32 P tile

  const int tid  = threadIdx.x;
  const int lane = tid & 31;
  const int wave = tid >> 5;
  const int tile = blockIdx.x * 8 + wave;   // 16-query-row tile id
  const int bh   = tile >> 7;               // 128 tiles per (b,h)
  const int q0   = (tile & 127) << 4;
  const int b    = bh >> 4;
  const int h    = bh & 15;
  const int l15  = lane & 15;
  const int half = lane >> 4;
  const int kb8  = half * 8;

  const _Float16* qrow = Q + ((size_t)bh * S_ + q0 + l15) * DK_;
  v16h qa0 = make_a16(*(const v8h*)(qrow + kb8),      *(const v8h*)(qrow + 16 + kb8));
  v16h qa1 = make_a16(*(const v8h*)(qrow + 32 + kb8), *(const v8h*)(qrow + 48 + kb8));

  float mrun[8], lrun[8];
  v8f zero = {};
  v8f oacc[4];
#pragma unroll
  for (int r = 0; r < 8; ++r) { mrun[r] = -INFINITY; lrun[r] = 0.0f; }
#pragma unroll
  for (int t = 0; t < 4; ++t) oacc[t] = zero;

  const float scale = 0.125f;  // 1/sqrt(64)
  const unsigned char* mb = mask + (size_t)b * S_ * S_;

  for (int kb = 0; kb < S_; kb += 32) {
    v8f s0 = zero, s1 = zero;
#pragma unroll
    for (int t = 0; t < 2; ++t) {
      const _Float16* krow = Kf + ((size_t)bh * S_ + kb + t * 16 + l15) * DK_;
      v16h kf0 = make_a16(*(const v8h*)(krow + kb8),      *(const v8h*)(krow + 16 + kb8));
      v16h kf1 = make_a16(*(const v8h*)(krow + 32 + kb8), *(const v8h*)(krow + 48 + kb8));
      v8f acc = t ? s1 : s0;
      acc = __builtin_amdgcn_wmma_f32_16x16x32_f16(false, qa0, false, kf0, (short)0, acc, false, false);
      acc = __builtin_amdgcn_wmma_f32_16x16x32_f16(false, qa1, false, kf1, (short)0, acc, false, false);
      if (t) s1 = acc; else s0 = acc;
    }

#pragma unroll
    for (int r = 0; r < 8; ++r) {
      const int q = q0 + r + 8 * half;
      float x0 = s0[r] * scale;
      float x1 = s1[r] * scale;
      if (mb[(size_t)q * S_ + kb + l15])      x0 = -INFINITY;
      if (mb[(size_t)q * S_ + kb + 16 + l15]) x1 = -INFINITY;

      float rmax = fmaxf(x0, x1);
#pragma unroll
      for (int off = 8; off >= 1; off >>= 1)
        rmax = fmaxf(rmax, __shfl_xor(rmax, off, 16));

      float mnew  = fmaxf(mrun[r], rmax);
      bool  dead  = (mnew == -INFINITY);
      float alpha = dead ? 1.0f : __expf(mrun[r] - mnew);
      float e0    = dead ? 0.0f : __expf(x0 - mnew);
      float e1    = dead ? 0.0f : __expf(x1 - mnew);

      float rs = e0 + e1;
#pragma unroll
      for (int off = 8; off >= 1; off >>= 1)
        rs += __shfl_xor(rs, off, 16);

      lrun[r] = lrun[r] * alpha + rs;
      mrun[r] = mnew;
#pragma unroll
      for (int t = 0; t < 4; ++t) oacc[t][r] = oacc[t][r] * alpha;

      ldsP[wave][r + 8 * half][l15]      = (_Float16)e0;
      ldsP[wave][r + 8 * half][16 + l15] = (_Float16)e1;
    }
    asm volatile("s_wait_dscnt 0x0" ::: "memory");  // same-wave DS in order

    v16h pa = frag_from_row(&ldsP[wave][l15][0], kb8);
#pragma unroll
    for (int t = 0; t < 4; ++t) {
      const _Float16* vcol = Vt + ((size_t)bh * DK_ + t * 16 + l15) * S_ + kb;
      v16h bv = make_a16(*(const v8h*)(vcol + kb8), *(const v8h*)(vcol + 16 + kb8));
      oacc[t] = __builtin_amdgcn_wmma_f32_16x16x32_f16(false, pa, false, bv, (short)0, oacc[t], false, false);
    }
  }

#pragma unroll
  for (int t = 0; t < 4; ++t) {
    const int dk = t * 16 + l15;
#pragma unroll
    for (int r = 0; r < 8; ++r) {
      const int q = q0 + r + 8 * half;
      float l = lrun[r];
      float v = (l > 0.0f) ? oacc[t][r] / l : 0.0f;
      O[((size_t)b * S_ + q) * D_ + h * DK_ + dk] = (_Float16)v;
    }
  }
}

extern "C" void kernel_launch(void* const* d_in, const int* in_sizes, int n_in,
                              void* d_out, int out_size, void* d_ws, size_t ws_size,
                              hipStream_t stream) {
  (void)in_sizes; (void)n_in; (void)out_size; (void)ws_size;
  const float* query = (const float*)d_in[0];
  const float* key   = (const float*)d_in[1];
  const float* value = (const float*)d_in[2];
  const unsigned char* mask = (const unsigned char*)d_in[3];
  const float* Wq = (const float*)d_in[4];
  const float* bq = (const float*)d_in[5];
  const float* Wk = (const float*)d_in[6];
  const float* bk = (const float*)d_in[7];
  const float* Wv = (const float*)d_in[8];
  const float* bv = (const float*)d_in[9];
  const float* Wo = (const float*)d_in[10];
  const float* bo = (const float*)d_in[11];

  char* ws = (char*)d_ws;
  const size_t seg = (size_t)M_ * D_ * sizeof(_Float16);  // 8 MiB each
  _Float16* qh = (_Float16*)(ws);                // later reused as Ah
  _Float16* kh = (_Float16*)(ws + seg);
  _Float16* vh = (_Float16*)(ws + 2 * seg);
  _Float16* Qh = (_Float16*)(ws + 3 * seg);
  _Float16* Kh = (_Float16*)(ws + 4 * seg);
  _Float16* Vt = (_Float16*)(ws + 5 * seg);
  _Float16* Ah = qh;  // qh is dead once the Q projection is done

  cvt_f32_to_f16<<<dim3((M_ * D_ / 4) / 256, 3), 256, 0, stream>>>(
      query, key, value, qh, kh, vh);

  dim3 ggrid(N_ / BN, M_ / BM);  // (8, 32)
  mha_gemm_wmma<1><<<ggrid, 256, 0, stream>>>(qh, Wq, bq, Qh);
  mha_gemm_wmma<1><<<ggrid, 256, 0, stream>>>(kh, Wk, bk, Kh);
  mha_gemm_wmma<2><<<ggrid, 256, 0, stream>>>(vh, Wv, bv, Vt);

  const int tiles = B_ * H_ * (S_ / 16);  // 4096 query tiles, 8 waves/block
  mha_flash_wmma<<<dim3(tiles / 8), 256, 0, stream>>>(Qh, Kh, Vt, mask, Ah);

  mha_gemm_wmma<0><<<ggrid, 256, 0, stream>>>(Ah, Wo, bo, d_out);
}
